// WindowAttention_9637906612585
// MI455X (gfx1250) — compile-verified, matrix operand order checked by reading, not run on hardware
//
#include <hip/hip_runtime.h>

// ---------------------------------------------------------------------------
// Fused shifted-window attention for MI455X (gfx1250, wave32, WMMA).
// One workgroup (256 threads = 8 waves) per 8x8 window; all intermediates in
// LDS (~72 KB -> 2 workgroups per WGP). All GEMMs use v_wmma_f32_16x16x32_bf16.
// Weights are pre-converted to bf16 once into d_ws by a small kernel, so the
// main kernel's WMMA B-operands are raw b64 global loads (L2-resident).
// ---------------------------------------------------------------------------

typedef __attribute__((ext_vector_type(16))) __bf16 v16bf;
typedef __attribute__((ext_vector_type(8)))  float  v8f;

union ABu { v16bf v; uint2 q[4]; __bf16 h[16]; };

// geometry
constexpr int N    = 64;    // pixels per window (8x8)
constexpr int CH   = 128;   // channels
// LDS strides (bf16 elements); padded for bank-conflict-free operand fetch
constexpr int STR  = 132;   // xs / qs / ks / os row stride
constexpr int PSTR = 68;    // probs row stride
constexpr int VSTR = 68;    // v-transposed row stride
// LDS layout (bf16 element offsets)
constexpr int SZ_A   = N * STR;                 // 8448  : xs | os (union)
constexpr int SZ_QK  = 4 * N * PSTR;            // 17408 : qs+ks | probs (union)
constexpr int OFF_QK = SZ_A;
constexpr int OFF_VT = SZ_A + SZ_QK;            // 25856
constexpr int ELS_BF = OFF_VT + CH * VSTR;      // 34560 bf16 elements
constexpr int BYTES_F = ELS_BF * 2;             // 69120 bytes
constexpr int SMEM_BYTES = BYTES_F + (512 + 228) * 4;  // + rn[512], pe[228] => 72080

constexpr int WQKV_ELS = 384 * 128;   // 49152
constexpr int WOUT_ELS = 128 * 128;   // 16384

__device__ __forceinline__ v8f vzero() {
  v8f z;
#pragma unroll
  for (int i = 0; i < 8; ++i) z[i] = 0.0f;
  return z;
}

__device__ __forceinline__ v8f wmma_bf16(v16bf a, v16bf b, v8f c) {
  return __builtin_amdgcn_wmma_f32_16x16x32_bf16(false, a, false, b, (short)0, c,
                                                 false, false);
}

// Load a 16x32 (or 32x16) bf16 WMMA operand from a row-major LDS array.
// lane<16 -> row=lane, K = kb+{0..7, 16..23}; lane>=16 -> K shifted by +8.
__device__ __forceinline__ v16bf lds_ab(const __bf16* base, int stride, int r0,
                                        int kb, int lane) {
  int r = r0 + (lane & 15);
  int k = kb + ((lane & 16) >> 1);
  const __bf16* p = base + r * stride + k;
  ABu u;
  u.q[0] = *(const uint2*)(p);
  u.q[1] = *(const uint2*)(p + 4);
  u.q[2] = *(const uint2*)(p + 16);
  u.q[3] = *(const uint2*)(p + 20);
  return u.v;
}

// Same operand fetch from a row-major bf16 weight matrix in global memory.
__device__ __forceinline__ v16bf ldg_b16(const __bf16* __restrict__ W, int ld,
                                         int n0, int kb, int lane) {
  int r = n0 + (lane & 15);
  int k = kb + ((lane & 16) >> 1);
  const __bf16* p = W + r * ld + k;
  ABu u;
  u.q[0] = *(const uint2*)(p);
  u.q[1] = *(const uint2*)(p + 4);
  u.q[2] = *(const uint2*)(p + 16);
  u.q[3] = *(const uint2*)(p + 20);
  return u.v;
}

// ---- one-shot weight conversion: fp32 -> bf16 into workspace ---------------
__global__ void __launch_bounds__(256)
cvt_weights_kernel(const float* __restrict__ wqkv, const float* __restrict__ wout,
                   __bf16* __restrict__ dst) {
  int g = (blockIdx.x * 256 + threadIdx.x) * 4;  // 4 elements per thread
  const float* src = (g < WQKV_ELS) ? (wqkv + g) : (wout + (g - WQKV_ELS));
  float4 f = *(const float4*)src;
  ABu t;
  t.h[0] = (__bf16)f.x; t.h[1] = (__bf16)f.y;
  t.h[2] = (__bf16)f.z; t.h[3] = (__bf16)f.w;
  *(uint2*)(dst + g) = t.q[0];
}

__global__ void __launch_bounds__(256)
swin_attn_kernel(const float* __restrict__ x, const __bf16* __restrict__ wq_bf,
                 const __bf16* __restrict__ wo_bf, const float* __restrict__ b_out,
                 const float* __restrict__ pos_emb, float* __restrict__ out) {
  extern __shared__ __align__(16) char smraw[];
  __bf16* sA  = (__bf16*)smraw;          // xs (phase 1)  | os (phase 5/6)
  __bf16* sQK = sA + OFF_QK;             // qs,ks (1..2)  | probs (4..5)
  __bf16* sVT = sA + OFF_VT;             // v transposed [ch][p]
  float*  rn  = (float*)(smraw + BYTES_F);  // [512]: 1/|q_row|, 1/|k_row|
  float*  pe  = rn + 512;                   // pos_emb copy [225]

  const int tid  = threadIdx.x;
  const int lane = tid & 31;
  const int wv   = tid >> 5;
  const int bx   = blockIdx.x;
  const int b    = bx >> 8;
  const int win  = bx & 255;
  const int wi   = win >> 4, wj = win & 15;
  const bool isTB = (wi == 15), isLR = (wj == 15);
  const int hi8  = (lane & 16) ? 8 : 0;

  // ---- P0: stage shifted x window to LDS (fp32 -> bf16); copy pos_emb ------
  {
    int p = tid >> 2, qtr = tid & 3;
    int r = p >> 3, c = p & 7;
    int ii = (wi * 8 + r + 4) & 127, jj = (wj * 8 + c + 4) & 127;
    const float* src = x + (((long)b * 128 + ii) * 128 + jj) * 128 + qtr * 32;
    __bf16* dst = sA + p * STR + qtr * 32;
#pragma unroll
    for (int e = 0; e < 8; ++e) {
      float4 f = *(const float4*)(src + e * 4);
      ABu t;
      t.h[0] = (__bf16)f.x; t.h[1] = (__bf16)f.y;
      t.h[2] = (__bf16)f.z; t.h[3] = (__bf16)f.w;
      *(uint2*)(dst + e * 4) = t.q[0];
    }
    for (int t2 = tid; t2 < 225; t2 += 256) pe[t2] = pos_emb[t2];
  }
  __syncthreads();

  // ---- P1: QKV GEMM 64x384x128. Each wave owns one M-tile and 12 N-tiles;
  //          A-operand (4 ksteps) hoisted and held in registers. ------------
  {
    int mt = wv >> 1, half = wv & 1;
    v16bf av[4];
#pragma unroll
    for (int ks = 0; ks < 4; ++ks) av[ks] = lds_ab(sA, STR, mt * 16, ks * 32, lane);
    for (int n = 0; n < 12; ++n) {
      int nt = half * 12 + n;
      v8f acc = vzero();
#pragma unroll
      for (int ks = 0; ks < 4; ++ks)
        acc = wmma_bf16(av[ks], ldg_b16(wq_bf, 128, nt * 16, ks * 32, lane), acc);
      int colL = lane & 15;
      if (nt < 16) {  // q (nt 0..7) or k (nt 8..15): row-major bf16
        __bf16* dst = sQK + (nt < 8 ? 0 : (N * STR));
        int ch = (nt & 7) * 16 + colL;
#pragma unroll
        for (int i = 0; i < 8; ++i)
          dst[(mt * 16 + i + hi8) * STR + ch] = (__bf16)acc[i];
      } else {        // v: store transposed [ch][p]; rows contiguous per lane
        int ch = (nt - 16) * 16 + colL;
        int p0 = mt * 16 + hi8;
        ABu tp;
#pragma unroll
        for (int i = 0; i < 8; ++i) tp.h[i] = (__bf16)acc[i];
        *(uint2*)(sVT + ch * VSTR + p0)     = tp.q[0];
        *(uint2*)(sVT + ch * VSTR + p0 + 4) = tp.q[1];
      }
    }
  }
  __syncthreads();

  // ---- P1.5: per-row inverse L2 norms of q and k (cosine attention) -------
  for (int t2 = tid; t2 < 512; t2 += 256) {
    int sel = t2 >> 8, rem = t2 & 255;
    int h = rem >> 6, row = rem & 63;
    const __bf16* p = sQK + sel * (N * STR) + row * STR + h * 32;
    ABu u0, u1;
#pragma unroll
    for (int e = 0; e < 4; ++e) {
      u0.q[e] = *(const uint2*)(p + e * 4);
      u1.q[e] = *(const uint2*)(p + 16 + e * 4);
    }
    float s = 0.0f;
#pragma unroll
    for (int e = 0; e < 16; ++e) {
      float a = (float)u0.h[e], bb = (float)u1.h[e];
      s += a * a + bb * bb;
    }
    rn[t2] = 1.0f / fmaxf(sqrtf(s), 1e-12f);
  }
  __syncthreads();

  // ---- P2: dots = q k^T per head (one WMMA per 16x16 tile, K=32) -----------
  v8f dacc[2][4];
#pragma unroll
  for (int g = 0; g < 2; ++g) {
    int grp = wv + g * 8, h = grp >> 2, mt = grp & 3;
    v16bf a = lds_ab(sQK, STR, mt * 16, h * 32, lane);
#pragma unroll
    for (int nt = 0; nt < 4; ++nt) {
      v16bf bm = lds_ab(sQK + N * STR, STR, nt * 16, h * 32, lane);
      dacc[g][nt] = wmma_bf16(a, bm, vzero());
    }
  }
  __syncthreads();  // all qs/ks reads done; region may be reused for probs

  // ---- P2b: scale + bias + shift-mask, register softmax, store probs ------
#pragma unroll
  for (int g = 0; g < 2; ++g) {
    int grp = wv + g * 8, h = grp >> 2, mt = grp & 3;
    int rbase = mt * 16 + hi8;
    float rq[8];
#pragma unroll
    for (int i = 0; i < 8; ++i) rq[i] = rn[h * 64 + rbase + i];
#pragma unroll
    for (int nt = 0; nt < 4; ++nt) {
      int j = nt * 16 + (lane & 15);
      float rk = rn[256 + h * 64 + j];
#pragma unroll
      for (int i = 0; i < 8; ++i) {
        int p = rbase + i;
        float v = dacc[g][nt][i] * rq[i] * rk * 50.0f  // 1/tau = 50
                + pe[((j >> 3) - (p >> 3) + 7) * 15 + ((j & 7) - (p & 7) + 7)];
        if (isTB && ((p >= 32) != (j >= 32)))           v = -__builtin_inff();
        if (isLR && (((p & 7) >= 4) != ((j & 7) >= 4))) v = -__builtin_inff();
        dacc[g][nt][i] = v;
      }
    }
#pragma unroll
    for (int i = 0; i < 8; ++i) {
      float mx = fmaxf(fmaxf(dacc[g][0][i], dacc[g][1][i]),
                       fmaxf(dacc[g][2][i], dacc[g][3][i]));
#pragma unroll
      for (int m = 1; m <= 8; m <<= 1) mx = fmaxf(mx, __shfl_xor(mx, m, 32));
      float sum = 0.0f;
#pragma unroll
      for (int nt = 0; nt < 4; ++nt) {
        float e = __expf(dacc[g][nt][i] - mx);
        dacc[g][nt][i] = e;
        sum += e;
      }
#pragma unroll
      for (int m = 1; m <= 8; m <<= 1) sum += __shfl_xor(sum, m, 32);
      float rcp = 1.0f / sum;
#pragma unroll
      for (int nt = 0; nt < 4; ++nt) dacc[g][nt][i] *= rcp;
    }
    __bf16* ph = sQK + h * (N * PSTR);
#pragma unroll
    for (int nt = 0; nt < 4; ++nt) {
      int j = nt * 16 + (lane & 15);
#pragma unroll
      for (int i = 0; i < 8; ++i)
        ph[(rbase + i) * PSTR + j] = (__bf16)dacc[g][nt][i];
    }
  }
  __syncthreads();

  // ---- P5: out = probs @ v (per head); write os row-major (aliases xs) -----
  {
    int h = wv >> 1;
    int mtb = (wv & 1) * 2;
    const __bf16* ph = sQK + h * (N * PSTR);
#pragma unroll
    for (int mm = 0; mm < 2; ++mm) {
      int mt = mtb + mm;
      v16bf a0 = lds_ab(ph, PSTR, mt * 16, 0, lane);
      v16bf a1 = lds_ab(ph, PSTR, mt * 16, 32, lane);
#pragma unroll
      for (int dt = 0; dt < 2; ++dt) {
        v8f acc = wmma_bf16(a0, lds_ab(sVT, VSTR, h * 32 + dt * 16, 0, lane), vzero());
        acc = wmma_bf16(a1, lds_ab(sVT, VSTR, h * 32 + dt * 16, 32, lane), acc);
        int ch = h * 32 + dt * 16 + (lane & 15);
#pragma unroll
        for (int i = 0; i < 8; ++i)
          sA[(mt * 16 + i + hi8) * STR + ch] = (__bf16)acc[i];
      }
    }
  }
  __syncthreads();

  // ---- P6: output projection + bias + reverse cyclic shift -----------------
  {
    int mt = wv >> 1, half = wv & 1;
    v16bf av[4];
#pragma unroll
    for (int ks = 0; ks < 4; ++ks) av[ks] = lds_ab(sA, STR, mt * 16, ks * 32, lane);
#pragma unroll
    for (int q = 0; q < 4; ++q) {
      int nt = half * 4 + q;
      v8f acc = vzero();
#pragma unroll
      for (int ks = 0; ks < 4; ++ks)
        acc = wmma_bf16(av[ks], ldg_b16(wo_bf, 128, nt * 16, ks * 32, lane), acc);
      int n = nt * 16 + (lane & 15);
      float bo = b_out[n];
#pragma unroll
      for (int i = 0; i < 8; ++i) {
        int p = mt * 16 + i + hi8;
        int r = p >> 3, c = p & 7;
        int ii = (wi * 8 + r + 4) & 127, jj = (wj * 8 + c + 4) & 127;
        out[(((long)b * 128 + ii) * 128 + jj) * 128 + n] = acc[i] + bo;
      }
    }
  }
}

extern "C" void kernel_launch(void* const* d_in, const int* in_sizes, int n_in,
                              void* d_out, int out_size, void* d_ws, size_t ws_size,
                              hipStream_t stream) {
  const float* x       = (const float*)d_in[0];
  const float* w_qkv   = (const float*)d_in[1];
  const float* w_out   = (const float*)d_in[2];
  const float* b_out   = (const float*)d_in[3];
  const float* pos_emb = (const float*)d_in[4];
  float* out = (float*)d_out;

  (void)in_sizes; (void)n_in; (void)out_size; (void)ws_size;

  __bf16* w_bf  = (__bf16*)d_ws;          // [WQKV_ELS + WOUT_ELS] bf16
  const __bf16* wq_bf = (const __bf16*)w_bf;
  const __bf16* wo_bf = wq_bf + WQKV_ELS;

  // 1) convert weights fp32 -> bf16 into workspace (cheap, every call;
  //    deterministic and graph-capture safe).
  cvt_weights_kernel<<<dim3((WQKV_ELS + WOUT_ELS) / (256 * 4)), dim3(256), 0,
                       stream>>>(w_qkv, w_out, w_bf);

  // 2) fused attention: one workgroup per (batch, window); ~72 KB dynamic LDS.
  hipFuncSetAttribute((const void*)swin_attn_kernel,
                      hipFuncAttributeMaxDynamicSharedMemorySize, SMEM_BYTES);
  swin_attn_kernel<<<dim3(4096), dim3(256), SMEM_BYTES, stream>>>(
      x, wq_bf, wo_bf, b_out, pos_emb, out);
}
